// CustomGRU_11175504904489
// MI455X (gfx1250) — matrix operand order repaired and checked by reading, not compiled, not run on hardware
//
#include <hip/hip_runtime.h>
#include <hip/hip_bf16.h>

// ---------------------------------------------------------------------------
// Bidirectional GRU, S=512, B=128, I=H=1024, fp32 in/out, bf16 WMMA compute.
//
//   1) pack_w   : Wih_f/Wih_b/Whh_f/Whh_b (f32 [3072,1024]) -> bf16 WMMA
//                 B-fragment layout [jt(192)][kk(32)][lane(32)][16 bf16]
//   2) pack_x   : inpt (f32 [S*B,1024]) -> bf16 WMMA A-fragment layout
//   3) zero     : h0 f32 buffer + h0 packed-fragment buffer
//   4) gi_gemm  : gi = x @ Wih^T + bih, both dirs, one big WMMA GEMM.
//                 K-loop FULLY unrolled: all fragment loads are
//                 base+immediate-offset, no loop-carried registers, scheduler
//                 freely overlaps load clauses with WMMAs.
//   5) gru_step : 512 launches. Reads h_prev as pre-packed bf16 A-fragments
//                 (produced by the previous step), does gh = h_prev @ Whh^T
//                 via WMMA for both dirs + fused gates, writes f32 output AND
//                 the packed A-fragments for the next step (ping-pong buffer).
//                 K-loop also fully unrolled.
// ---------------------------------------------------------------------------

typedef __bf16 bf16_t;
typedef __attribute__((ext_vector_type(16))) __bf16 v16bf;
typedef __attribute__((ext_vector_type(8)))  float  v8f;

static constexpr int S  = 512;
static constexpr int B  = 128;
static constexpr int I  = 1024;
static constexpr int H  = 1024;
static constexpr int H3 = 3 * H;           // 3072
static constexpr int OW = 2 * H;           // 2048, output row width
static constexpr int KT = I / 32;          // 32 k-steps of depth 32
static constexpr int NT_W = H3 / 16;       // 192 column tiles in a weight matrix
static constexpr int RT_X = (S * B) / 16;  // 4096 row tiles of X
static constexpr int JT_H = H / 16;        // 64 column tiles of hidden state
// Packed h fragments: [dir(2)][bm(8)][kk(32)][lane(32)][e(16)] bf16
static constexpr size_t HPK_ELEMS = (size_t)2 * 8 * 32 * 32 * 16;   // 262144

__device__ inline v8f wmma_bf16(v16bf a, v16bf b, v8f c) {
  return __builtin_amdgcn_wmma_f32_16x16x32_bf16(
      false, a, false, b, (short)0, c, false, false);
}

__device__ inline float sigmoidf_(float x) { return 1.0f / (1.0f + __expf(-x)); }

// ---------------------------------------------------------------------------
// Pack a [3072,1024] f32 weight matrix into WMMA B-fragment order.
// B (32x16, 16-bit): lanes 0-15 -> col N=lane, K=kk*32+{0..7,16..23};
// lanes 16-31 -> col N=lane-16, K=kk*32+{8..15,24..31}.
// ---------------------------------------------------------------------------
__global__ void pack_w(const float* __restrict__ W, bf16_t* __restrict__ out) {
  int s = blockIdx.x * 256 + threadIdx.x;       // slot: 192*32*32 = 196608
  int lane = s & 31;
  int kk   = (s >> 5) & 31;
  int jt   = s >> 10;
  int row  = jt * 16 + (lane & 15);
  int k0   = kk * 32 + ((lane < 16) ? 0 : 8);
  const float* src = W + (size_t)row * I + k0;
  bf16_t* dst = out + (size_t)s * 16;
#pragma unroll
  for (int e = 0; e < 16; ++e) dst[e] = (bf16_t)src[e < 8 ? e : e + 8];
}

// ---------------------------------------------------------------------------
// Pack X [S*B,1024] f32 into WMMA A-fragment order.
// A (16x32, 16-bit): lanes 0-15: M=lane, K=kk*32+{0..7,16..23};
// lanes 16-31: M=lane-16, K=kk*32+{8..15,24..31}.
// ---------------------------------------------------------------------------
__global__ void pack_x(const float* __restrict__ X, bf16_t* __restrict__ out) {
  int s = blockIdx.x * 256 + threadIdx.x;       // slot: 4096*32*32 = 4194304
  int lane = s & 31;
  int kk   = (s >> 5) & 31;
  int rt   = s >> 10;
  const float* src = X + ((size_t)rt * 16 + (lane & 15)) * I;
  int k0 = kk * 32 + ((lane < 16) ? 0 : 8);
  int k1 = k0 + 16;
  bf16_t* dst = out + (size_t)s * 16;
#pragma unroll
  for (int e = 0; e < 8; ++e) dst[e]     = (bf16_t)src[k0 + e];
#pragma unroll
  for (int e = 0; e < 8; ++e) dst[8 + e] = (bf16_t)src[k1 + e];
}

__global__ void zero_buf(float* __restrict__ p, int n) {
  int i = blockIdx.x * 256 + threadIdx.x;
  if (i < n) p[i] = 0.0f;
}

// ---------------------------------------------------------------------------
// gi = X @ Wih^T + bih, both directions. Wave owns a 16x64 output tile.
// Fully unrolled K loop: immediate-offset loads, maximal load/WMMA overlap.
// ---------------------------------------------------------------------------
__global__ __launch_bounds__(256)
void gi_gemm(const bf16_t* __restrict__ xpk,
             const bf16_t* __restrict__ wih_pk_f, const bf16_t* __restrict__ wih_pk_b,
             const float*  __restrict__ bih_f,    const float*  __restrict__ bih_b,
             float* __restrict__ gi_f, float* __restrict__ gi_b) {
  const int lane = threadIdx.x & 31;
  const int wid  = blockIdx.x * 8 + (threadIdx.x >> 5);
  const int dir  = wid / (RT_X * 48);
  const int rem  = wid % (RT_X * 48);
  const int rt   = rem / 48;
  const int ng   = rem % 48;

  const bf16_t* wpk  = dir ? wih_pk_b : wih_pk_f;
  const float*  bih  = dir ? bih_b    : bih_f;
  float*        gout = dir ? gi_b     : gi_f;

  // Lane base pointers; everything below is immediate offsets from these.
  const bf16_t* abase = xpk + (size_t)rt * KT * 32 * 16 + (size_t)lane * 16;
  const bf16_t* bbase = wpk + (size_t)(ng * 4) * KT * 32 * 16 + (size_t)lane * 16;

  v8f acc[4] = {v8f{}, v8f{}, v8f{}, v8f{}};
#pragma unroll
  for (int kk = 0; kk < KT; ++kk) {
    v16bf a = *(const v16bf*)(abase + (size_t)kk * 32 * 16);
#pragma unroll
    for (int u = 0; u < 4; ++u) {
      v16bf b = *(const v16bf*)(bbase + ((size_t)u * KT + kk) * 32 * 16);
      acc[u] = wmma_bf16(a, b, acc[u]);
    }
  }

  const int col   = lane & 15;
  const int mbase = (lane >= 16) ? 8 : 0;
#pragma unroll
  for (int u = 0; u < 4; ++u) {
    int j = (ng * 4 + u) * 16 + col;
    float bv = bih[j];
#pragma unroll
    for (int i = 0; i < 8; ++i) {
      int row = rt * 16 + mbase + i;
      gout[(size_t)row * H3 + j] = acc[u][i] + bv;
    }
  }
}

// ---------------------------------------------------------------------------
// One GRU timestep, both directions. Grid: dim3(8 cg, 8 bm, 2 dir), 256 thr.
// Wave w -> hidden col tile jt = cg*8 + w; block covers 128 hidden cols,
// which is exactly k-fragment groups kk = cg*4 .. cg*4+3 of the NEXT step.
// Reads h_prev as pre-packed bf16 A-fragments (no staging, no pre-barrier);
// writes f32 output and next-step packed fragments (ping-pong).
// ---------------------------------------------------------------------------
__global__ __launch_bounds__(256)
void gru_step(const bf16_t* __restrict__ whh_pk_f, const bf16_t* __restrict__ whh_pk_b,
              const float*  __restrict__ gi_f_t,   const float*  __restrict__ gi_b_t,
              const float*  __restrict__ hprev_f,  const float*  __restrict__ hprev_b,
              const bf16_t* __restrict__ hpk_in,   bf16_t* __restrict__ hpk_out,
              const float*  __restrict__ bhh_f,    const float*  __restrict__ bhh_b,
              float* __restrict__ out_t) {
  __shared__ __align__(32) bf16_t lds_h[16 * 128];   // 4 KB transpose tile

  const int dir  = blockIdx.z;
  const int bm   = blockIdx.y;
  const int cg   = blockIdx.x;
  const int wave = threadIdx.x >> 5;
  const int lane = threadIdx.x & 31;
  const int jt   = cg * 8 + wave;               // hidden col tile 0..63

  const bf16_t* wpk = dir ? whh_pk_b : whh_pk_f;
  // Lane base pointers; gate g lives at immediate offset g*JT_H*KT*32*16
  // (2 MB in bytes) from bbase, kk at kk*32*16 -- all within 24-bit IOFFSET.
  const bf16_t* abase = hpk_in + ((size_t)dir * 8 + bm) * KT * 32 * 16
                               + (size_t)lane * 16;
  const bf16_t* bbase = wpk + (size_t)jt * KT * 32 * 16 + (size_t)lane * 16;

  v8f ar{}, az{}, an{};
#pragma unroll
  for (int kk = 0; kk < KT; ++kk) {
    v16bf a  = *(const v16bf*)(abase + (size_t)kk * 32 * 16);
    v16bf br = *(const v16bf*)(bbase + ((size_t)(0 * JT_H) * KT + kk) * 32 * 16);
    v16bf bz = *(const v16bf*)(bbase + ((size_t)(1 * JT_H) * KT + kk) * 32 * 16);
    v16bf bn = *(const v16bf*)(bbase + ((size_t)(2 * JT_H) * KT + kk) * 32 * 16);
    ar = wmma_bf16(a, br, ar);
    az = wmma_bf16(a, bz, az);
    an = wmma_bf16(a, bn, an);
  }

  // Gates. acc lane holds col j = jt*16 + (lane&15), rows mbase..mbase+7.
  const float* gi  = dir ? gi_b_t  : gi_f_t;    // [B,3H] f32, bih folded in
  const float* hp  = dir ? hprev_b : hprev_f;   // f32, row stride OW
  const float* bhh = dir ? bhh_b   : bhh_f;
  const int col    = lane & 15;
  const int j      = jt * 16 + col;             // hidden index 0..1023
  const int mbase  = (lane >= 16) ? 8 : 0;
  const int lcol   = wave * 16 + col;           // col within block's 128
  const float bhr = bhh[j], bhz = bhh[H + j], bhn = bhh[2 * H + j];
  const int outcol = dir ? (H + j) : j;

#pragma unroll
  for (int i = 0; i < 8; ++i) {
    int row = bm * 16 + mbase + i;              // batch index
    size_t gbase = (size_t)row * H3;
    float gir = gi[gbase + j];
    float giz = gi[gbase + H + j];
    float gin = gi[gbase + 2 * H + j];
    float hprev = hp[(size_t)row * OW + j];
    float r = sigmoidf_(gir + ar[i] + bhr);
    float z = sigmoidf_(giz + az[i] + bhz);
    float n = tanhf(gin + r * (an[i] + bhn));
    float hn = (1.0f - z) * n + z * hprev;
    out_t[(size_t)row * OW + outcol] = hn;
    lds_h[(mbase + i) * 128 + lcol] = (bf16_t)hn;   // stage for transpose
  }
  __syncthreads();

  // Emit next-step A-fragments for this block's 128 hidden cols
  // (fragment groups kk = cg*4 .. cg*4+3 of rows bm*16..bm*16+15).
  if (threadIdx.x < 128) {
    int kk_l = threadIdx.x >> 5;                // 0..3
    int ln   = threadIdx.x & 31;
    int m    = ln & 15;
    int kb   = kk_l * 32 + ((ln < 16) ? 0 : 8); // col base within block
    const bf16_t* srow = lds_h + m * 128;
    bf16_t* dst = hpk_out +
        ((((size_t)dir * 8 + bm) * KT + (cg * 4 + kk_l)) * 32 + ln) * 16;
#pragma unroll
    for (int e = 0; e < 8; ++e) dst[e]     = srow[kb + e];
#pragma unroll
    for (int e = 0; e < 8; ++e) dst[8 + e] = srow[kb + 16 + e];
  }
}

// ---------------------------------------------------------------------------
// Launch
// ---------------------------------------------------------------------------
extern "C" void kernel_launch(void* const* d_in, const int* in_sizes, int n_in,
                              void* d_out, int out_size, void* d_ws, size_t ws_size,
                              hipStream_t stream) {
  const float* inpt  = (const float*)d_in[0];
  const float* Wih_f = (const float*)d_in[1];
  const float* Whh_f = (const float*)d_in[2];
  const float* bih_f = (const float*)d_in[3];
  const float* bhh_f = (const float*)d_in[4];
  const float* Wih_b = (const float*)d_in[5];
  const float* Whh_b = (const float*)d_in[6];
  const float* bih_b = (const float*)d_in[7];
  const float* bhh_b = (const float*)d_in[8];
  float* out = (float*)d_out;

  char* ws = (char*)d_ws;
  size_t off = 0;
  bf16_t* xpk      = (bf16_t*)(ws + off); off += (size_t)RT_X * KT * 32 * 16 * 2;  // 128 MB
  bf16_t* wih_pk_f = (bf16_t*)(ws + off); off += (size_t)NT_W * KT * 32 * 16 * 2;  // 6 MB
  bf16_t* wih_pk_b = (bf16_t*)(ws + off); off += (size_t)NT_W * KT * 32 * 16 * 2;
  bf16_t* whh_pk_f = (bf16_t*)(ws + off); off += (size_t)NT_W * KT * 32 * 16 * 2;
  bf16_t* whh_pk_b = (bf16_t*)(ws + off); off += (size_t)NT_W * KT * 32 * 16 * 2;
  bf16_t* hpk0     = (bf16_t*)(ws + off); off += HPK_ELEMS * 2;                    // 512 KB
  bf16_t* hpk1     = (bf16_t*)(ws + off); off += HPK_ELEMS * 2;                    // 512 KB
  float*  zero_h   = (float*) (ws + off); off += (size_t)B * OW * 4;               // 1 MB
  float*  gi_f     = (float*) (ws + off); off += (size_t)S * B * H3 * 4;           // 805 MB
  float*  gi_b     = (float*) (ws + off); off += (size_t)S * B * H3 * 4;           // 805 MB
  (void)ws_size;

  // 1) Pack weights (L2-resident for the whole scan).
  pack_w<<<768, 256, 0, stream>>>(Wih_f, wih_pk_f);
  pack_w<<<768, 256, 0, stream>>>(Wih_b, wih_pk_b);
  pack_w<<<768, 256, 0, stream>>>(Whh_f, whh_pk_f);
  pack_w<<<768, 256, 0, stream>>>(Whh_b, whh_pk_b);

  // 2) Pack X.
  pack_x<<<RT_X * 32 * 32 / 256, 256, 0, stream>>>(inpt, xpk);

  // 3) Zero h0 (f32 row buffer + packed-fragment buffer 0).
  zero_buf<<<(B * OW + 255) / 256, 256, 0, stream>>>(zero_h, B * OW);
  zero_buf<<<((int)(HPK_ELEMS / 2) + 255) / 256, 256, 0, stream>>>(
      (float*)hpk0, (int)(HPK_ELEMS / 2));

  // 4) All input projections, both directions, one big WMMA GEMM.
  gi_gemm<<<2 * RT_X * 48 / 8, 256, 0, stream>>>(xpk, wih_pk_f, wih_pk_b,
                                                 bih_f, bih_b, gi_f, gi_b);

  // 5) Sequential scan: 512 step kernels (fwd + bwd fused). Backward reads
  //    gi of x[S-1-t] and stores outputs in produced (reversed) order,
  //    matching the reference exactly.
  for (int t = 0; t < S; ++t) {
    const float* gi_f_t  = gi_f + (size_t)t * B * H3;
    const float* gi_b_t  = gi_b + (size_t)(S - 1 - t) * B * H3;
    const float* hprev_f = (t == 0) ? zero_h     : (out + (size_t)(t - 1) * B * OW);
    const float* hprev_b = (t == 0) ? zero_h + H : (out + (size_t)(t - 1) * B * OW + H);
    const bf16_t* hin  = (t & 1) ? hpk1 : hpk0;
    bf16_t*       hout = (t & 1) ? hpk0 : hpk1;
    float* out_t = out + (size_t)t * B * OW;
    gru_step<<<dim3(8, 8, 2), 256, 0, stream>>>(whh_pk_f, whh_pk_b,
                                                gi_f_t, gi_b_t,
                                                hprev_f, hprev_b,
                                                hin, hout,
                                                bhh_f, bhh_b, out_t);
  }
}
